// ScalEdgesWrite_22651657519352
// MI455X (gfx1250) — matrix operand order repaired and checked by reading, not compiled, not run on hardware
//
#include <hip/hip_runtime.h>

typedef __attribute__((ext_vector_type(16))) _Float16 v16h;
typedef __attribute__((ext_vector_type(8)))  _Float16 v8h;
typedef __attribute__((ext_vector_type(2)))  __fp16   v2hf;   // cvt_pkrtz result type
typedef __attribute__((ext_vector_type(8)))  float    v8f;
typedef __attribute__((ext_vector_type(4)))  float    v4f;

#define D_DIM 128
#define INV_SQRT2 0.70710678118654752f

// ---------------------------------------------------------------------------
// Fragment-packed LDS layout for W (f16).
// 64 fragment blocks: fb = proj*32 + nt*4 + kc   (proj in {0,1}, nt 0..7, kc 0..3)
// Each block: 32 lanes x 8 DWORDs (v16h B-fragment for v_wmma_f32_16x16x32_f16).
// Lanes 0..15 (h=0) in [0, 8192) DWORDs, lanes 16..31 (h=1) in [8192, 16384)
// with a +4 DWORD rotation so each ds_load_b128 touches every bank exactly
// twice (minimum for 128 DWORDs). Address is affine in fb:
//   dword_addr = lane_base{A,B} + fb*128   -> fb part folds into DS imm offset.
// ---------------------------------------------------------------------------
__device__ __forceinline__ int frag_dword(int fb, int lane, int d) {
  const int h = lane >> 4;
  const int n = lane & 15;
  const int idx = n * 8 + d;
  return h ? (8192 + fb * 128 + ((idx + 4) & 127))
           : (fb * 128 + idx);
}

__device__ __forceinline__ v16h load_bfrag(const unsigned* __restrict__ baseA,
                                           const unsigned* __restrict__ baseB,
                                           int fb128 /* = fb*128, literal */) {
  const v8h lo = *(const v8h*)(baseA + fb128);
  const v8h hi = *(const v8h*)(baseB + fb128);
  v16h r;
#pragma unroll
  for (int i = 0; i < 8; ++i) { r[i] = lo[i]; r[8 + i] = hi[i]; }
  return r;
}

// A-fragment (16x32 f16, ISA layout): lane m = lane&15 is the row; h = lane>>4.
// halves 0..7  <- K = kbase +      h*8 .. +7   (VGPR0-3)
// halves 8..15 <- K = kbase + 16 + h*8 .. +7   (VGPR4-7)
__device__ __forceinline__ v16h load_afrag(const float* __restrict__ xrow, int kc, int h) {
  const float* p = xrow + kc * 32 + h * 8;
  const v4f c0 = *(const v4f*)(p);
  const v4f c1 = *(const v4f*)(p + 4);
  const v4f c2 = *(const v4f*)(p + 16);
  const v4f c3 = *(const v4f*)(p + 20);
  const v2hf p0 = __builtin_amdgcn_cvt_pkrtz(c0[0], c0[1]);
  const v2hf p1 = __builtin_amdgcn_cvt_pkrtz(c0[2], c0[3]);
  const v2hf p2 = __builtin_amdgcn_cvt_pkrtz(c1[0], c1[1]);
  const v2hf p3 = __builtin_amdgcn_cvt_pkrtz(c1[2], c1[3]);
  const v2hf p4 = __builtin_amdgcn_cvt_pkrtz(c2[0], c2[1]);
  const v2hf p5 = __builtin_amdgcn_cvt_pkrtz(c2[2], c2[3]);
  const v2hf p6 = __builtin_amdgcn_cvt_pkrtz(c3[0], c3[1]);
  const v2hf p7 = __builtin_amdgcn_cvt_pkrtz(c3[2], c3[3]);
  v16h a;
  a[0]  = p0[0]; a[1]  = p0[1]; a[2]  = p1[0]; a[3]  = p1[1];
  a[4]  = p2[0]; a[5]  = p2[1]; a[6]  = p3[0]; a[7]  = p3[1];
  a[8]  = p4[0]; a[9]  = p4[1]; a[10] = p5[0]; a[11] = p5[1];
  a[12] = p6[0]; a[13] = p6[1]; a[14] = p7[0]; a[15] = p7[1];
  return a;
}

// SADDR-form fp32 atomic add with constant instruction offset:
//   global_atomic_add_f32 v_off, v_data, s[base] offset:IMM scope:SCOPE_DEV
// 32-bit VGPR byte offset + SGPR-pair base; no "memory" clobber so unrelated
// LDS/VMEM loads can be scheduled around the atomic stream.
template <int IMM>
__device__ __forceinline__ void atomic_add_saddr(float* __restrict__ base,
                                                 unsigned byte_off, float v) {
  asm volatile("global_atomic_add_f32 %0, %1, %2 offset:%3 scope:SCOPE_DEV"
               :: "v"(byte_off), "v"(v), "s"(base), "n"(IMM));
}

__global__ void __launch_bounds__(256)
zero_out_kernel(float* __restrict__ out, int n) {
  int i = (blockIdx.x * 256 + threadIdx.x) * 4;
  if (i + 3 < n) {
    *(v4f*)(out + i) = (v4f){0.f, 0.f, 0.f, 0.f};
  } else {
    for (int j = i; j < n; ++j) out[j] = 0.f;
  }
}

// ---------------------------------------------------------------------------
// Fused dual-projection GEMM + scaled atomic scatter.
// Rows flattened g in [0, R), R = B*E; b = g/E, e = g%E.
// out[(b*N + src[e])*128 + o] += inv_sqrt2*coeff[src[e]] * (x[g] . W_src[o])
// out[(b*N + dst[e])*128 + o] += inv_sqrt2*coeff[dst[e]] * (x[g] . W_dst[o])
// ---------------------------------------------------------------------------
__global__ void __launch_bounds__(256)
fused_proj_scatter_kernel(const float* __restrict__ x,
                          const long long* __restrict__ src,
                          const long long* __restrict__ dst,
                          const float* __restrict__ coeff,
                          const float* __restrict__ Ws,
                          const float* __restrict__ Wd,
                          float* __restrict__ out,
                          unsigned R, unsigned E, unsigned N) {
  __shared__ unsigned ldsW[16384];  // 64 KB of packed f16 B-fragments

  // ---- stage W_src / W_dst into fragment-packed LDS (once per WG) ----
  for (int q = threadIdx.x; q < 2048; q += 256) {   // 64 blocks * 32 lanes
    const int fb = q >> 5, lq = q & 31;
    const int proj = fb >> 5, ntq = (fb >> 2) & 7, kcq = fb & 3;
    const int hq = lq >> 4, nq = lq & 15;
    const float* Wrow = (proj ? Wd : Ws) + (ntq * 16 + nq) * D_DIM + kcq * 32 + hq * 8;
    float wf[16];
    *(v4f*)(wf)      = *(const v4f*)(Wrow);
    *(v4f*)(wf + 4)  = *(const v4f*)(Wrow + 4);
    *(v4f*)(wf + 8)  = *(const v4f*)(Wrow + 16);
    *(v4f*)(wf + 12) = *(const v4f*)(Wrow + 20);
#pragma unroll
    for (int d = 0; d < 8; ++d) {
      const v2hf pk = __builtin_amdgcn_cvt_pkrtz(wf[2 * d], wf[2 * d + 1]);
      ldsW[frag_dword(fb, lq, d)] = __builtin_bit_cast(unsigned, pk);
    }
  }
  __syncthreads();

  const int lane = threadIdx.x & 31;
  const int wave = threadIdx.x >> 5;
  const int h = lane >> 4;
  const int n = lane & 15;

  // Per-lane LDS fragment base pointers (fb part goes into DS imm offset).
  const unsigned* baseA = ldsW + (h ? (8192 + ((n * 8 + 4) & 127)) : (n * 8));
  const unsigned* baseB = ldsW + (h ? (8192 + ((n * 8 + 8) & 127)) : (n * 8 + 4));

  const unsigned nTiles  = (R + 31) >> 5;           // 32 rows per wave-iteration
  const unsigned wstride = gridDim.x * 8;

  for (unsigned t = blockIdx.x * 8 + wave; t < nTiles; t += wstride) {
    const unsigned base = t * 32;

    // ---- A fragments: 2 M-tiles x 4 K-chunks (fp32 -> f16 in regs) ----
    v16h A[2][4];
#pragma unroll
    for (int mt = 0; mt < 2; ++mt) {
      unsigned g = base + mt * 16 + n;
      if (g >= R) g = R - 1;                        // clamp; masked by scale=0
      const float* xrow = x + (size_t)g * D_DIM;
#pragma unroll
      for (int kc = 0; kc < 4; ++kc) A[mt][kc] = load_afrag(xrow, kc, h);
    }

    // ---- per-row scatter metadata: 32-bit BYTE offsets (incl. n*4) + scale ----
    unsigned voffS[2][8], voffD[2][8];
    float    sS[2][8],    sD[2][8];
#pragma unroll
    for (int mt = 0; mt < 2; ++mt) {
#pragma unroll
      for (int r = 0; r < 8; ++r) {
        unsigned g = base + mt * 16 + r + 8 * h;    // D-matrix row for this lane
        const float live = (g < R) ? INV_SQRT2 : 0.0f;
        if (g >= R) g = R - 1;
        const unsigned b = g / E;
        const unsigned e = g - b * E;
        const unsigned ns = (unsigned)src[e];
        const unsigned nd = (unsigned)dst[e];
        voffS[mt][r] = (b * N + ns) * (D_DIM * 4u) + n * 4u;
        voffD[mt][r] = (b * N + nd) * (D_DIM * 4u) + n * 4u;
        sS[mt][r] = live * coeff[ns];
        sD[mt][r] = live * coeff[nd];
      }
    }

    // ---- 8 N-tiles (literal nt): WMMA over K=128, scaled atomic scatter ----
    // DS address = per-lane base + literal imm; atomic col part = literal imm.
#define NT_BLOCK(NT)                                                           \
    {                                                                          \
      v8f aS0 = {}, aS1 = {}, aD0 = {}, aD1 = {};                              \
      _Pragma("unroll")                                                        \
      for (int kc = 0; kc < 4; ++kc) {                                         \
        const v16h bS = load_bfrag(baseA, baseB, ((NT) * 4 + kc) * 128);       \
        aS0 = __builtin_amdgcn_wmma_f32_16x16x32_f16(false, A[0][kc], false,   \
                 bS, (short)0, aS0, false, false);                             \
        aS1 = __builtin_amdgcn_wmma_f32_16x16x32_f16(false, A[1][kc], false,   \
                 bS, (short)0, aS1, false, false);                             \
      }                                                                        \
      _Pragma("unroll")                                                        \
      for (int kc = 0; kc < 4; ++kc) {                                         \
        const v16h bD = load_bfrag(baseA, baseB, (32 + (NT) * 4 + kc) * 128);  \
        aD0 = __builtin_amdgcn_wmma_f32_16x16x32_f16(false, A[0][kc], false,   \
                 bD, (short)0, aD0, false, false);                             \
        aD1 = __builtin_amdgcn_wmma_f32_16x16x32_f16(false, A[1][kc], false,   \
                 bD, (short)0, aD1, false, false);                             \
      }                                                                        \
      _Pragma("unroll")                                                        \
      for (int r = 0; r < 8; ++r) {                                            \
        atomic_add_saddr<(NT) * 64>(out, voffS[0][r], aS0[r] * sS[0][r]);      \
        atomic_add_saddr<(NT) * 64>(out, voffS[1][r], aS1[r] * sS[1][r]);      \
        atomic_add_saddr<(NT) * 64>(out, voffD[0][r], aD0[r] * sD[0][r]);      \
        atomic_add_saddr<(NT) * 64>(out, voffD[1][r], aD1[r] * sD[1][r]);      \
      }                                                                        \
    }
    NT_BLOCK(0) NT_BLOCK(1) NT_BLOCK(2) NT_BLOCK(3)
    NT_BLOCK(4) NT_BLOCK(5) NT_BLOCK(6) NT_BLOCK(7)
#undef NT_BLOCK
  }
}

extern "C" void kernel_launch(void* const* d_in, const int* in_sizes, int n_in,
                              void* d_out, int out_size, void* d_ws, size_t ws_size,
                              hipStream_t stream) {
  const float*     x     = (const float*)d_in[0];
  const long long* src   = (const long long*)d_in[1];
  const long long* dst   = (const long long*)d_in[2];
  const float*     coeff = (const float*)d_in[3];
  const float*     Ws    = (const float*)d_in[4];
  const float*     Wd    = (const float*)d_in[5];
  float*           out   = (float*)d_out;

  const unsigned E = (unsigned)in_sizes[1];
  const unsigned N = (unsigned)in_sizes[3];
  const unsigned R = (unsigned)((size_t)in_sizes[0] / D_DIM);  // B*E rows

  const int zgrid = (out_size / 4 + 255) / 256;
  zero_out_kernel<<<zgrid, 256, 0, stream>>>(out, out_size);

  fused_proj_scatter_kernel<<<512, 256, 0, stream>>>(x, src, dst, coeff, Ws, Wd,
                                                     out, R, E, N);
}
